// TopCache_52192442581891
// MI455X (gfx1250) — compile-verified, matrix operand order checked by reading, not run on hardware
//
#include <hip/hip_runtime.h>
#include <hip/hip_bf16.h>
#include <math.h>

#define VSIZE   8192
#define NTOK    8192          // B*S = 8*1024
#define TOPK    32
#define CK      64
#define BETA    0.9f
#define MINP    0.5f
#define NEG_INF (-1e30f)

typedef __attribute__((ext_vector_type(16))) _Float16 v16h;
typedef __attribute__((ext_vector_type(8)))  float    v8f;

#if defined(__AMDGCN__) && __has_builtin(__builtin_amdgcn_global_load_async_to_lds_b32)
#  define HAVE_ASYNC_LDS 1
#else
#  define HAVE_ASYNC_LDS 0
#endif

__device__ __forceinline__ float wred_sum(float v) {
#pragma unroll
  for (int o = 16; o > 0; o >>= 1) v += __shfl_xor(v, o, 32);
  return v;
}
__device__ __forceinline__ float wred_max(float v) {
#pragma unroll
  for (int o = 16; o > 0; o >>= 1) v = fmaxf(v, __shfl_xor(v, o, 32));
  return v;
}

// ---------------------------------------------------------------- init
__global__ void init_kernel(float* loss, int* counts) {
  int i = blockIdx.x * blockDim.x + threadIdx.x;
  if (i == 0) *loss = 0.0f;
  if (i < VSIZE) counts[i] = 0;
}

// ---------------------------------------------------- per-token kernel
// 1 block (256 thr) per token: KL loss + destructive top-32 of x row.
__global__ __launch_bounds__(256)
void tok_kernel(const float* __restrict__ x, const int* __restrict__ gold,
                const int* __restrict__ cache_index, const float* __restrict__ cache_p,
                float* __restrict__ loss_out, int* __restrict__ counts,
                int* __restrict__ topi_ws, float* __restrict__ newp_ws) {
  __shared__ float xs[VSIZE];        // 32 KB row buffer (fits easily: 320KB/WGP)
  __shared__ float rv[256];
  __shared__ int   ri[256];
  __shared__ float s_topv[TOPK];
  __shared__ int   s_topi[TOPK];

  const int t   = blockIdx.x;
  const int tid = threadIdx.x;
  const float* xrow = x + (size_t)t * VSIZE;

  // ---- stream x row into LDS (async-to-LDS path when available) ----
#if HAVE_ASYNC_LDS
  for (int i = tid; i < VSIZE; i += 256) {
    __builtin_amdgcn_global_load_async_to_lds_b32(
        (__attribute__((address_space(1))) int*)(xrow + i),
        (__attribute__((address_space(3))) int*)(xs + i), 0, 0);
  }
#if __has_builtin(__builtin_amdgcn_s_wait_asynccnt)
  __builtin_amdgcn_s_wait_asynccnt(0);
#else
  asm volatile("s_wait_asynccnt 0" ::: "memory");
#endif
#else
  for (int i = tid; i < VSIZE; i += 256) xs[i] = xrow[i];
#endif
  __syncthreads();

  const int  g     = gold[t];
  const bool valid = (g != 0);       // gold_pad_mask = (gold == 0)

  // ---- KL loss on wave 0 (32 lanes == 32 cache entries) ----
  if (tid < 32) {
    const int   j   = tid;
    float       cp  = cache_p[(size_t)g * CK + j];
    const int   ci  = cache_index[(size_t)g * CK + j];
    const float cps = wred_sum(cp);
    float       cpn = cp / cps;
    if (!valid) cpn = 0.0f;
    const float ms  = xs[ci];
    const float mx  = wred_max(ms);
    const float se  = wred_sum(__expf(ms - mx));
    const float lgp = ms - mx - __logf(se);
    const float kl  = (cpn > 0.0f) ? cpn * (__logf(cpn) - lgp) : 0.0f;
    const float ks  = wred_sum(kl);
    if (j == 0 && ks != 0.0f) atomicAdd(loss_out, ks);
    if (j == 0 && valid) atomicAdd(counts + g, 1);
  }
  __syncthreads();

  // ---- iterative top-32 (destroys xs) ----
  for (int r = 0; r < TOPK; ++r) {
    float bv = NEG_INF; int bi = 0;
    for (int i = tid; i < VSIZE; i += 256) {
      float v = xs[i];
      if (v > bv) { bv = v; bi = i; }
    }
    rv[tid] = bv; ri[tid] = bi;
    __syncthreads();
    for (int s = 128; s > 0; s >>= 1) {
      if (tid < s && rv[tid + s] > rv[tid]) { rv[tid] = rv[tid + s]; ri[tid] = ri[tid + s]; }
      __syncthreads();
    }
    if (tid == 0) { s_topv[r] = rv[0]; s_topi[r] = ri[0]; xs[ri[0]] = NEG_INF; }
    __syncthreads();
  }

  // ---- gold insertion + softmax + MIN_GOLD_P fix (wave 0) ----
  if (tid < 32) {
    const int j = tid;
    float v = s_topv[j];
    int   i = s_topi[j];
    unsigned long long found = __ballot(i == g);
    if (found == 0ull && j == TOPK - 1) i = g;   // force gold into last slot
    const bool  gm = (i == g);
    const float mx = wred_max(v);
    const float e  = __expf(v - mx);
    const float se = wred_sum(e);
    float       p  = e / se;
    const float gp = wred_sum(gm ? p : 0.0f);
    if (gp < MINP) {
      const float scale = (1.0f - MINP) / fmaxf(1.0f - gp, 1e-12f);
      p = gm ? MINP : p * scale;
    }
    newp_ws[(size_t)t * TOPK + j] = valid ? (1.0f - BETA) * p : 0.0f;
    topi_ws[(size_t)t * TOPK + j] = i;
  }
}

// ------------------------------------------------------ per-row kernel
// 1 block (256 thr) per vocab row: sparse acc in a dense 32KB LDS row,
// then LDS top-64.  Avoids the reference's 256MB dense acc entirely.
__global__ __launch_bounds__(256)
void row_kernel(const int* __restrict__ gold,
                const int* __restrict__ cache_index, const float* __restrict__ cache_p,
                const int* __restrict__ counts,
                const int* __restrict__ topi_ws, const float* __restrict__ newp_ws,
                int* __restrict__ cand_cols, float* __restrict__ cand_vals) {
  __shared__ float acc[VSIZE];       // dense per-row accumulator: 32 KB of LDS
  __shared__ float rv[256];
  __shared__ int   ri[256];

  const int r   = blockIdx.x;
  const int tid = threadIdx.x;
  const int cnt = counts[r];
  if (cnt == 0) return;              // row not updated -> finalize copies old cache

  for (int i = tid; i < VSIZE; i += 256) acc[i] = 0.0f;
  __syncthreads();

  // decayed old entries (ds_add_f32 merges duplicate columns)
  for (int k = tid; k < CK; k += 256) {
    const int c = cache_index[(size_t)r * CK + k];
    atomicAdd(&acc[c], BETA * cache_p[(size_t)r * CK + k]);
  }
  // new contributions from tokens whose gold == r
  const float inv = 1.0f / (float)(cnt > 1 ? cnt : 1);
  for (int t = tid; t < NTOK; t += 256) {
    __builtin_prefetch(&gold[(t + 256) < NTOK ? (t + 256) : t], 0, 1);
    const int gt = gold[t];
    if (gt == r && gt != 0) {
#pragma unroll 4
      for (int k = 0; k < TOPK; ++k) {
        atomicAdd(&acc[topi_ws[(size_t)t * TOPK + k]],
                  newp_ws[(size_t)t * TOPK + k] * inv);
      }
    }
  }
  __syncthreads();

  // top-64 over the LDS row (all candidates are >= 0; -1 is a safe sentinel)
  for (int s = 0; s < CK; ++s) {
    float bv = -1.0f; int bi = 0;
    for (int i = tid; i < VSIZE; i += 256) {
      float v = acc[i];
      if (v > bv) { bv = v; bi = i; }
    }
    rv[tid] = bv; ri[tid] = bi;
    __syncthreads();
    for (int w = 128; w > 0; w >>= 1) {
      if (tid < w && rv[tid + w] > rv[tid]) { rv[tid] = rv[tid + w]; ri[tid] = ri[tid + w]; }
      __syncthreads();
    }
    if (tid == 0) {
      cand_cols[(size_t)r * CK + s] = ri[0];
      cand_vals[(size_t)r * CK + s] = rv[0];
      acc[ri[0]] = -1.0f;
    }
    __syncthreads();
  }
}

// -------------------------------------------------------- finalize
// 1 wave per 16 rows. WMMA (f16 A x ones-B -> f32 C) computes 16 row sums
// of the 64 candidate values in one matrix op pair, then normalize+select.
__global__ __launch_bounds__(32)
void finalize_kernel(const int* __restrict__ cache_index, const float* __restrict__ cache_p,
                     const int* __restrict__ counts,
                     const int* __restrict__ cand_cols, const float* __restrict__ cand_vals,
                     float* __restrict__ out_ci, float* __restrict__ out_cp) {
  const int base = blockIdx.x * 16;           // rows base..base+15
  const int lane = threadIdx.x;
  const int m    = lane & 15;                 // A-matrix row = lane (per 16-bit A layout)
  const int kh   = (lane >> 4) & 1;           // upper-half lanes hold K+8 slices
  const int row  = base + m;
  const float sel = (counts[row] > 0) ? 1.0f : 0.0f;

  v16h a0, a1, bones;
  v8f  c = {};
#pragma unroll
  for (int e = 0; e < 16; ++e) {
    // 16-bit A 16x32 layout: lane<16 holds K = {0..7,16..23}; lane>=16 K = {8..15,24..31}
    const int k = e + ((e >= 8) ? 8 : 0) + (kh ? 8 : 0);
    a0[e] = (_Float16)(sel * cand_vals[(size_t)row * CK + k]);        // cols  0..31
    a1[e] = (_Float16)(sel * cand_vals[(size_t)row * CK + 32 + k]);   // cols 32..63
    bones[e] = (_Float16)1.0f;                                        // ones B => row sums
  }
  c = __builtin_amdgcn_wmma_f32_16x16x32_f16(false, a0, false, bones, (short)0, c, false, false);
  c = __builtin_amdgcn_wmma_f32_16x16x32_f16(false, a1, false, bones, (short)0, c, false, false);

  float carr[8];
#pragma unroll
  for (int i = 0; i < 8; ++i) carr[i] = c[i];

  // C layout: VGPR i @ lane 0 -> (M=i, N=0); VGPR i @ lane 16 -> (M=8+i, N=0)
  __shared__ float rowsum[16];
  if (lane == 0) {
#pragma unroll
    for (int i = 0; i < 8; ++i) rowsum[i] = carr[i];
  }
  if (lane == 16) {
#pragma unroll
    for (int i = 0; i < 8; ++i) rowsum[8 + i] = carr[i];
  }
  __syncthreads();

  for (int idx = lane; idx < 16 * CK; idx += 32) {
    const int rr  = base + idx / CK;
    const int k   = idx % CK;
    const bool upd = counts[rr] > 0;
    float civ, cpv;
    if (upd) {
      const float denom = fmaxf(rowsum[rr - base], 1e-12f);
      civ = (float)cand_cols[(size_t)rr * CK + k];
      cpv = cand_vals[(size_t)rr * CK + k] / denom;
    } else {
      civ = (float)cache_index[(size_t)rr * CK + k];
      cpv = cache_p[(size_t)rr * CK + k];
    }
    out_ci[(size_t)rr * CK + k] = civ;
    out_cp[(size_t)rr * CK + k] = cpv;
  }
}

// ---------------------------------------------------------------- launch
extern "C" void kernel_launch(void* const* d_in, const int* in_sizes, int n_in,
                              void* d_out, int out_size, void* d_ws, size_t ws_size,
                              hipStream_t stream) {
  const float* x           = (const float*)d_in[0];
  const int*   gold        = (const int*)d_in[1];
  // d_in[2] = gold_pad_mask (recomputed as gold==0 to avoid bool-dtype ambiguity)
  const int*   cache_index = (const int*)d_in[3];
  const float* cache_p     = (const float*)d_in[4];

  float* out     = (float*)d_out;            // [0]=loss, then ci, then cp
  float* out_ci  = out + 1;
  float* out_cp  = out + 1 + (size_t)VSIZE * CK;

  char* ws = (char*)d_ws;
  int*   counts = (int*)ws;                 ws += (size_t)VSIZE * sizeof(int);
  int*   topi   = (int*)ws;                 ws += (size_t)NTOK * TOPK * sizeof(int);
  float* newp   = (float*)ws;               ws += (size_t)NTOK * TOPK * sizeof(float);
  int*   ccols  = (int*)ws;                 ws += (size_t)VSIZE * CK * sizeof(int);
  float* cvals  = (float*)ws;

  init_kernel<<<(VSIZE + 255) / 256, 256, 0, stream>>>(out, counts);
  tok_kernel<<<NTOK, 256, 0, stream>>>(x, gold, cache_index, cache_p,
                                       out, counts, topi, newp);
  row_kernel<<<VSIZE, 256, 0, stream>>>(gold, cache_index, cache_p, counts,
                                        topi, newp, ccols, cvals);
  finalize_kernel<<<VSIZE / 16, 32, 0, stream>>>(cache_index, cache_p, counts,
                                                 ccols, cvals, out_ci, out_cp);
}
